// SparseLlamaAttention_79310866088657
// MI455X (gfx1250) — matrix operand order repaired and checked by reading, not compiled, not run on hardware
//
#include <hip/hip_runtime.h>
#include <hip/hip_bf16.h>

// ---------------------------------------------------------------------------
// Sparse (heavy-hitter) Llama attention for gfx1250 (MI455X).
// bf16 WMMA everywhere + async global->LDS tile staging (ASYNCcnt path).
// ---------------------------------------------------------------------------

typedef __attribute__((ext_vector_type(16))) __bf16 v16bf;
typedef __attribute__((ext_vector_type(8)))  float  v8f;

#define S_LEN 2048
#define HID_D 2048
#define NH    16
#define NKV   8
#define HD    128
#define QD    (NH * HD)    // 2048
#define KVD   (NKV * HD)   // 1024
#define HEAVY 1228
#define NEGF  (-3.3895313892515355e38f)   // bfloat16 finfo.min (matches jnp)
#define SCALE 0.08838834764831845f        // 1/sqrt(128)

union Frag16 { unsigned short u[16]; v16bf v; };

__device__ inline unsigned short f2bf(float f) {
  unsigned int u = __float_as_uint(f);
  unsigned int r = u + 0x7FFFu + ((u >> 16) & 1u);   // round-to-nearest-even
  return (unsigned short)(r >> 16);
}

__device__ inline v8f vzero8() { v8f z = {}; return z; }

__device__ inline v8f wmma_bf16(v16bf a, v16bf b, v8f c) {
  // D = A(16x32 bf16) * B(32x16 bf16) + C(16x16 f32)
  return __builtin_amdgcn_wmma_f32_16x16x32_bf16(false, a, false, b,
                                                 (short)0, c, false, false);
}

// ---- async global->LDS (CDNA5 GLOBAL_LOAD_ASYNC_TO_LDS_*, ASYNCcnt) -------
// LDS aperture occupies the high 32 bits of a flat address; the low 32 bits
// are the wave-relative LDS byte offset (ISA 10.2 aperture calc), so a
// truncated generic pointer is a valid VDST LDS address.
__device__ inline void async_copy16(void* lds_dst, const void* gsrc) {
  unsigned l = (unsigned)(unsigned long long)lds_dst;
  unsigned long long g = (unsigned long long)gsrc;
  asm volatile("global_load_async_to_lds_b128 %0, %1, off"
               :: "v"(l), "v"(g) : "memory");
}
__device__ inline void async_wait() {
  asm volatile("s_wait_asynccnt 0x0" ::: "memory");
}

// A fragment (16x32, bf16) from row-major source, ld in elements.
// element e of lane L: m = L&15, k = (L>>4)*16 + e
__device__ inline v16bf load_a(const unsigned short* p, int ld, int lane) {
  Frag16 f; const int half = lane >> 4, m = lane & 15;
#pragma unroll
  for (int e = 0; e < 16; ++e) {
    int k = half * 16 + e;
    f.u[e] = p[m * ld + k];
  }
  return f.v;
}
// B fragment (32x16) from row-major KxN source: k = (L>>4)*16 + e, n = L&15
__device__ inline v16bf load_b(const unsigned short* p, int ld, int lane) {
  Frag16 f; const int half = lane >> 4, n = lane & 15;
#pragma unroll
  for (int e = 0; e < 16; ++e) f.u[e] = p[(half * 16 + e) * ld + n];
  return f.v;
}
// B fragment from transposed source (N x K row-major, e.g. K-cache [key][hd])
__device__ inline v16bf load_bt(const unsigned short* p, int ld, int lane) {
  Frag16 f; const int half = lane >> 4, n = lane & 15;
#pragma unroll
  for (int e = 0; e < 16; ++e) f.u[e] = p[n * ld + half * 16 + e];
  return f.v;
}

// ---------------------------------------------------------------------------
// f32 -> bf16 convert (vectorized)
__global__ __launch_bounds__(256) void cvt_bf16_kernel(
    const float* __restrict__ in, unsigned short* __restrict__ out, int n4) {
  int i = blockIdx.x * 256 + threadIdx.x;
  if (i < n4) {
    float4 v = ((const float4*)in)[i];
    ushort4 o = make_ushort4(f2bf(v.x), f2bf(v.y), f2bf(v.z), f2bf(v.w));
    ((ushort4*)out)[i] = o;
  }
}

// ---------------------------------------------------------------------------
// Generic bf16 GEMM: C[MxN] = A[MxK] * B[KxN], f32 out. 128x128x32 tiles.
__global__ __launch_bounds__(256) void gemm_bf16_kernel(
    const unsigned short* __restrict__ A, const unsigned short* __restrict__ B,
    float* __restrict__ C, int M, int N, int K) {
  __shared__ unsigned short As[128][32];
  __shared__ unsigned short Bs[32][128];
  const int tid = threadIdx.x;
  const int lane = tid & 31, w = tid >> 5;
  const int wr = w >> 1, wc = w & 1;                 // wave: 32 rows x 64 cols
  const int bm = blockIdx.y * 128, bn = blockIdx.x * 128;

  v8f acc[2][4];
#pragma unroll
  for (int i = 0; i < 2; ++i)
#pragma unroll
    for (int j = 0; j < 4; ++j) acc[i][j] = vzero8();

  const int arow = tid >> 1, acol = (tid & 1) * 16;
  const int brow = tid >> 3, bcol = (tid & 7) * 16;

  for (int k0 = 0; k0 < K; k0 += 32) {
    const unsigned short* sa = A + (size_t)(bm + arow) * K + k0 + acol;
    async_copy16(&As[arow][acol],     sa);
    async_copy16(&As[arow][acol] + 8, sa + 8);
    const unsigned short* sb = B + (size_t)(k0 + brow) * N + bn + bcol;
    async_copy16(&Bs[brow][bcol],     sb);
    async_copy16(&Bs[brow][bcol] + 8, sb + 8);
    async_wait();
    __syncthreads();

    v16bf bf[4];
#pragma unroll
    for (int nt = 0; nt < 4; ++nt) bf[nt] = load_b(&Bs[0][wc * 64 + nt * 16], 128, lane);
#pragma unroll
    for (int mt = 0; mt < 2; ++mt) {
      v16bf a = load_a(&As[wr * 32 + mt * 16][0], 32, lane);
#pragma unroll
      for (int nt = 0; nt < 4; ++nt) acc[mt][nt] = wmma_bf16(a, bf[nt], acc[mt][nt]);
    }
    __syncthreads();
  }
  const int half = lane >> 4, ln = lane & 15;
#pragma unroll
  for (int mt = 0; mt < 2; ++mt)
#pragma unroll
    for (int nt = 0; nt < 4; ++nt)
#pragma unroll
      for (int r = 0; r < 8; ++r) {
        int m = bm + wr * 32 + mt * 16 + half * 8 + r;
        int n = bn + wc * 64 + nt * 16 + ln;
        C[(size_t)m * N + n] = acc[mt][nt][r];
      }
}

// ---------------------------------------------------------------------------
// RoPE + bf16 convert. One thread per (s, head, d<64) pair.
__global__ __launch_bounds__(256) void rope_cvt_kernel(
    const float* __restrict__ X, const int* __restrict__ pos,
    unsigned short* __restrict__ Out, int nheads) {
  int idx = blockIdx.x * 256 + threadIdx.x;
  int total = S_LEN * nheads * 64;
  if (idx >= total) return;
  int d = idx & 63;
  int t = idx >> 6;
  int hh = t % nheads;
  int s = t / nheads;
  float p = (float)pos[s];
  float inv = powf(10000.0f, -(float)d * (1.0f / 64.0f));
  float fr = p * inv;
  float c = cosf(fr), sn = sinf(fr);
  const float* row = X + (size_t)s * nheads * HD + hh * HD;
  float x1 = row[d], x2 = row[d + 64];
  unsigned short* orow = Out + (size_t)s * nheads * HD + hh * HD;
  orow[d]      = f2bf(x1 * c - x2 * sn);
  orow[d + 64] = f2bf(x2 * c + x1 * sn);
}

// ---------------------------------------------------------------------------
// Shared score-tile compute: 64(q) x 64(k) scores via WMMA, wave = 32x16.
__device__ inline void score_tile(const unsigned short (&Qs)[64][HD],
                                  const unsigned short (&Ks)[64][HD],
                                  int wr, int wc, int lane, v8f acc[2]) {
  acc[0] = vzero8(); acc[1] = vzero8();
#pragma unroll
  for (int kc = 0; kc < 4; ++kc) {                      // HD / 32
    v16bf b = load_bt(&Ks[wc * 16][kc * 32], HD, lane); // B = K^T
#pragma unroll
    for (int mt = 0; mt < 2; ++mt) {
      v16bf a = load_a(&Qs[wr * 32 + mt * 16][kc * 32], HD, lane);
      acc[mt] = wmma_bf16(a, b, acc[mt]);
    }
  }
}

// 64x128 bf16 tile: global -> LDS via async copies (64 B per thread).
__device__ inline void load_tile64(unsigned short (&T)[64][HD],
                                   const unsigned short* src, int rowStride, int tid) {
  int row = tid >> 2, seg = (tid & 3) * 32;
  const unsigned short* s = src + (size_t)row * rowStride + seg;
  unsigned short* d = &T[row][seg];
  async_copy16(d,      s);
  async_copy16(d + 8,  s + 8);
  async_copy16(d + 16, s + 16);
  async_copy16(d + 24, s + 24);
  async_wait();
}

// ---------------------------------------------------------------------------
// Pass 1: first softmax statistics + probability column-sum partials.
// grid = (S/64, NH). Deterministic (no atomics).
__global__ __launch_bounds__(256) void attn_stats_kernel(
    const unsigned short* __restrict__ Qb, const unsigned short* __restrict__ Kb,
    float* __restrict__ colpart) {
  __shared__ unsigned short Qs[64][HD];
  __shared__ unsigned short Ks[64][HD];
  __shared__ float red[64][4];
  __shared__ float rowmaxS[64];
  __shared__ float rowsumS[64];
  __shared__ float colAcc[S_LEN];
  __shared__ float colPart[64][4];

  const int tid = threadIdx.x;
  const int lane = tid & 31, w = tid >> 5;
  const int wr = w >> 2, wc = w & 3;       // wave: 32 rows x 16 key-cols
  const int half = lane >> 4, ln = lane & 15;
  const int h = blockIdx.y, kv = h >> 1;
  const int q0 = blockIdx.x * 64;
  const int NT = S_LEN / 64;

  load_tile64(Qs, Qb + (size_t)q0 * QD + h * HD, QD, tid);

  // -------- sweep 1: row max --------
  float rmx[2][8];
#pragma unroll
  for (int mt = 0; mt < 2; ++mt)
#pragma unroll
    for (int r = 0; r < 8; ++r) rmx[mt][r] = NEGF;

  for (int kt = 0; kt < NT; ++kt) {
    load_tile64(Ks, Kb + (size_t)(kt * 64) * KVD + kv * HD, KVD, tid);
    __syncthreads();
    v8f acc[2]; score_tile(Qs, Ks, wr, wc, lane, acc);
    int j = kt * 64 + wc * 16 + ln;
#pragma unroll
    for (int mt = 0; mt < 2; ++mt) {
      int i0 = q0 + wr * 32 + mt * 16 + half * 8;
#pragma unroll
      for (int r = 0; r < 8; ++r) {
        float s = acc[mt][r] * SCALE + ((j <= i0 + r) ? 0.0f : NEGF);
        rmx[mt][r] = fmaxf(rmx[mt][r], s);
      }
    }
    __syncthreads();
  }
#pragma unroll
  for (int mt = 0; mt < 2; ++mt)
#pragma unroll
    for (int r = 0; r < 8; ++r) {
      float v = rmx[mt][r];
      for (int off = 1; off < 16; off <<= 1) v = fmaxf(v, __shfl_xor(v, off, 32));
      rmx[mt][r] = v;
    }
  if (ln == 0)
#pragma unroll
    for (int mt = 0; mt < 2; ++mt)
#pragma unroll
      for (int r = 0; r < 8; ++r)
        red[wr * 32 + mt * 16 + half * 8 + r][wc] = rmx[mt][r];
  __syncthreads();
  if (tid < 64)
    rowmaxS[tid] = fmaxf(fmaxf(red[tid][0], red[tid][1]), fmaxf(red[tid][2], red[tid][3]));
  __syncthreads();

  float rm[2][8];
#pragma unroll
  for (int mt = 0; mt < 2; ++mt)
#pragma unroll
    for (int r = 0; r < 8; ++r) rm[mt][r] = rowmaxS[wr * 32 + mt * 16 + half * 8 + r];

  // -------- sweep 2: row sum of exp --------
  float rs[2][8];
#pragma unroll
  for (int mt = 0; mt < 2; ++mt)
#pragma unroll
    for (int r = 0; r < 8; ++r) rs[mt][r] = 0.0f;

  for (int kt = 0; kt < NT; ++kt) {
    load_tile64(Ks, Kb + (size_t)(kt * 64) * KVD + kv * HD, KVD, tid);
    __syncthreads();
    v8f acc[2]; score_tile(Qs, Ks, wr, wc, lane, acc);
    int j = kt * 64 + wc * 16 + ln;
#pragma unroll
    for (int mt = 0; mt < 2; ++mt) {
      int i0 = q0 + wr * 32 + mt * 16 + half * 8;
#pragma unroll
      for (int r = 0; r < 8; ++r) {
        float s = acc[mt][r] * SCALE + ((j <= i0 + r) ? 0.0f : NEGF);
        rs[mt][r] += expf(s - rm[mt][r]);
      }
    }
    __syncthreads();
  }
#pragma unroll
  for (int mt = 0; mt < 2; ++mt)
#pragma unroll
    for (int r = 0; r < 8; ++r) {
      float v = rs[mt][r];
      for (int off = 1; off < 16; off <<= 1) v += __shfl_xor(v, off, 32);
      rs[mt][r] = v;
    }
  if (ln == 0)
#pragma unroll
    for (int mt = 0; mt < 2; ++mt)
#pragma unroll
      for (int r = 0; r < 8; ++r)
        red[wr * 32 + mt * 16 + half * 8 + r][wc] = rs[mt][r];
  __syncthreads();
  if (tid < 64)
    rowsumS[tid] = (red[tid][0] + red[tid][1]) + (red[tid][2] + red[tid][3]);
  __syncthreads();

  float rsi[2][8];
#pragma unroll
  for (int mt = 0; mt < 2; ++mt)
#pragma unroll
    for (int r = 0; r < 8; ++r)
      rsi[mt][r] = 1.0f / rowsumS[wr * 32 + mt * 16 + half * 8 + r];

  // -------- sweep 3: probability column sums (fixed-order, deterministic) ---
  for (int kt = 0; kt < NT; ++kt) {
    load_tile64(Ks, Kb + (size_t)(kt * 64) * KVD + kv * HD, KVD, tid);
    __syncthreads();
    v8f acc[2]; score_tile(Qs, Ks, wr, wc, lane, acc);
    int j = kt * 64 + wc * 16 + ln;
    float colp = 0.0f;
#pragma unroll
    for (int mt = 0; mt < 2; ++mt) {
      int i0 = q0 + wr * 32 + mt * 16 + half * 8;
#pragma unroll
      for (int r = 0; r < 8; ++r) {
        float s = acc[mt][r] * SCALE + ((j <= i0 + r) ? 0.0f : NEGF);
        colp += expf(s - rm[mt][r]) * rsi[mt][r];
      }
    }
    colPart[wc * 16 + ln][wr * 2 + half] = colp;
    __syncthreads();
    if (tid < 64)
      colAcc[kt * 64 + tid] =
          (colPart[tid][0] + colPart[tid][1]) + (colPart[tid][2] + colPart[tid][3]);
    __syncthreads();
  }
  for (int jj = tid; jj < S_LEN; jj += 256)
    colpart[((size_t)blockIdx.x * NH + h) * S_LEN + jj] = colAcc[jj];
}

// Reduce per-block column-sum partials in fixed order.
__global__ __launch_bounds__(256) void colsum_reduce_kernel(
    const float* __restrict__ colpart, float* __restrict__ colsum) {
  int idx = blockIdx.x * 256 + threadIdx.x;   // over NH * S
  int h = idx >> 11, jj = idx & (S_LEN - 1);
  float s = 0.0f;
  for (int qt = 0; qt < S_LEN / 64; ++qt)
    s += colpart[((size_t)qt * NH + h) * S_LEN + jj];
  colsum[idx] = s;
}

// ---------------------------------------------------------------------------
// Per-head top-k threshold via in-LDS bitonic sort (ascending).
__global__ __launch_bounds__(256) void topk_thresh_kernel(
    const float* __restrict__ colsum, float* __restrict__ thresh) {
  __shared__ float arr[S_LEN];
  const int h = blockIdx.x;
  for (int i = threadIdx.x; i < S_LEN; i += 256) arr[i] = colsum[h * S_LEN + i];
  __syncthreads();
  for (int k = 2; k <= S_LEN; k <<= 1) {
    for (int j = k >> 1; j > 0; j >>= 1) {
      for (int i = threadIdx.x; i < S_LEN; i += 256) {
        int ixj = i ^ j;
        if (ixj > i) {
          float a = arr[i], b = arr[ixj];
          bool up = ((i & k) == 0);
          if (up ? (a > b) : (a < b)) { arr[i] = b; arr[ixj] = a; }
        }
      }
      __syncthreads();
    }
  }
  if (threadIdx.x == 0) thresh[h] = arr[S_LEN - HEAVY];  // HEAVY-th largest
}

// ---------------------------------------------------------------------------
// Pass 2: masked softmax (heavy mask, causal implicit) + P@V, bf16 output.
__global__ __launch_bounds__(256) void attn_out_kernel(
    const unsigned short* __restrict__ Qb, const unsigned short* __restrict__ Kb,
    const unsigned short* __restrict__ Vb, const float* __restrict__ colsum,
    const float* __restrict__ thresh, unsigned short* __restrict__ Ob) {
  __shared__ unsigned short Qs[64][HD];
  __shared__ unsigned short Ks[64][HD];   // shared between K tile and V tile
  __shared__ unsigned short Ps[64][64];
  __shared__ float red[64][4];
  __shared__ float rowmaxS[64];
  __shared__ float rowsumS[64];
  __shared__ float keepf[64];

  const int tid = threadIdx.x;
  const int lane = tid & 31, w = tid >> 5;
  const int wr = w >> 2, wc = w & 3;
  const int half = lane >> 4, ln = lane & 15;
  const int h = blockIdx.y, kv = h >> 1;
  const int q0 = blockIdx.x * 64;
  const int NT = S_LEN / 64;
  const float th = thresh[h];

  load_tile64(Qs, Qb + (size_t)q0 * QD + h * HD, QD, tid);

  // -------- sweep 1: masked row max --------
  float rmx[2][8];
#pragma unroll
  for (int mt = 0; mt < 2; ++mt)
#pragma unroll
    for (int r = 0; r < 8; ++r) rmx[mt][r] = NEGF;

  for (int kt = 0; kt < NT; ++kt) {
    if (tid < 64)
      keepf[tid] = (colsum[h * S_LEN + kt * 64 + tid] >= th) ? 1.0f : 0.0f;
    load_tile64(Ks, Kb + (size_t)(kt * 64) * KVD + kv * HD, KVD, tid);
    __syncthreads();
    v8f acc[2]; score_tile(Qs, Ks, wr, wc, lane, acc);
    int jl = wc * 16 + ln;
    int j = kt * 64 + jl;
    float kf = keepf[jl];
#pragma unroll
    for (int mt = 0; mt < 2; ++mt) {
      int i0 = q0 + wr * 32 + mt * 16 + half * 8;
#pragma unroll
      for (int r = 0; r < 8; ++r) {
        float base = acc[mt][r] * SCALE + ((j <= i0 + r) ? 0.0f : NEGF);
        float s = (kf != 0.0f) ? base : NEGF;
        rmx[mt][r] = fmaxf(rmx[mt][r], s);
      }
    }
    __syncthreads();
  }
#pragma unroll
  for (int mt = 0; mt < 2; ++mt)
#pragma unroll
    for (int r = 0; r < 8; ++r) {
      float v = rmx[mt][r];
      for (int off = 1; off < 16; off <<= 1) v = fmaxf(v, __shfl_xor(v, off, 32));
      rmx[mt][r] = v;
    }
  if (ln == 0)
#pragma unroll
    for (int mt = 0; mt < 2; ++mt)
#pragma unroll
      for (int r = 0; r < 8; ++r)
        red[wr * 32 + mt * 16 + half * 8 + r][wc] = rmx[mt][r];
  __syncthreads();
  if (tid < 64)
    rowmaxS[tid] = fmaxf(fmaxf(red[tid][0], red[tid][1]), fmaxf(red[tid][2], red[tid][3]));
  __syncthreads();
  float rm[2][8];
#pragma unroll
  for (int mt = 0; mt < 2; ++mt)
#pragma unroll
    for (int r = 0; r < 8; ++r) rm[mt][r] = rowmaxS[wr * 32 + mt * 16 + half * 8 + r];

  // -------- sweep 2: masked row sum --------
  float rsum[2][8];
#pragma unroll
  for (int mt = 0; mt < 2; ++mt)
#pragma unroll
    for (int r = 0; r < 8; ++r) rsum[mt][r] = 0.0f;

  for (int kt = 0; kt < NT; ++kt) {
    if (tid < 64)
      keepf[tid] = (colsum[h * S_LEN + kt * 64 + tid] >= th) ? 1.0f : 0.0f;
    load_tile64(Ks, Kb + (size_t)(kt * 64) * KVD + kv * HD, KVD, tid);
    __syncthreads();
    v8f acc[2]; score_tile(Qs, Ks, wr, wc, lane, acc);
    int jl = wc * 16 + ln;
    int j = kt * 64 + jl;
    float kf = keepf[jl];
#pragma unroll
    for (int mt = 0; mt < 2; ++mt) {
      int i0 = q0 + wr * 32 + mt * 16 + half * 8;
#pragma unroll
      for (int r = 0; r < 8; ++r) {
        float base = acc[mt][r] * SCALE + ((j <= i0 + r) ? 0.0f : NEGF);
        float s = (kf != 0.0f) ? base : NEGF;
        rsum[mt][r] += expf(s - rm[mt][r]);
      }
    }
    __syncthreads();
  }
#pragma unroll
  for (int mt = 0; mt < 2; ++mt)
#pragma unroll
    for (int r = 0; r < 8; ++r) {
      float v = rsum[mt][r];
      for (int off = 1; off < 16; off <<= 1) v += __shfl_xor(v, off, 32);
      rsum[mt][r] = v;
    }
  if (ln == 0)
#pragma unroll
    for (int mt = 0; mt < 2; ++mt)
#pragma unroll
      for (int r = 0; r < 8; ++r)
        red[wr * 32 + mt * 16 + half * 8 + r][wc] = rsum[mt][r];
  __syncthreads();
  if (tid < 64)
    rowsumS[tid] = (red[tid][0] + red[tid][1]) + (red[tid][2] + red[tid][3]);
  __syncthreads();
  float rsi[2][8];
#pragma unroll
  for (int mt = 0; mt < 2; ++mt)
#pragma unroll
    for (int r = 0; r < 8; ++r)
      rsi[mt][r] = 1.0f / rowsumS[wr * 32 + mt * 16 + half * 8 + r];

  // -------- sweep 3: P tile -> LDS, then P @ V accumulation --------
  v8f oacc[2][2];
#pragma unroll
  for (int mt = 0; mt < 2; ++mt)
#pragma unroll
    for (int nt = 0; nt < 2; ++nt) oacc[mt][nt] = vzero8();

  for (int kt = 0; kt < NT; ++kt) {
    if (tid < 64)
      keepf[tid] = (colsum[h * S_LEN + kt * 64 + tid] >= th) ? 1.0f : 0.0f;
    load_tile64(Ks, Kb + (size_t)(kt * 64) * KVD + kv * HD, KVD, tid);
    __syncthreads();
    {
      v8f acc[2]; score_tile(Qs, Ks, wr, wc, lane, acc);
      int jl = wc * 16 + ln;
      int j = kt * 64 + jl;
      float kf = keepf[jl];
#pragma unroll
      for (int mt = 0; mt < 2; ++mt) {
        int i0 = q0 + wr * 32 + mt * 16 + half * 8;
#pragma unroll
        for (int r = 0; r < 8; ++r) {
          float base = acc[mt][r] * SCALE + ((j <= i0 + r) ? 0.0f : NEGF);
          float s = (kf != 0.0f) ? base : NEGF;
          float p = expf(s - rm[mt][r]) * rsi[mt][r];
          Ps[wr * 32 + mt * 16 + half * 8 + r][jl] = f2bf(p);
        }
      }
    }
    __syncthreads();
    load_tile64(Ks, Vb + (size_t)(kt * 64) * KVD + kv * HD, KVD, tid);  // V tile
    __syncthreads();
#pragma unroll
    for (int kc = 0; kc < 2; ++kc) {          // 64 keys = 2 x K32
      v16bf bf[2];
#pragma unroll
      for (int nt = 0; nt < 2; ++nt)
        bf[nt] = load_b(&Ks[kc * 32][(w & 3) * 32 + nt * 16], HD, lane);
#pragma unroll
      for (int mt = 0; mt < 2; ++mt) {
        v16bf a = load_a(&Ps[wr * 32 + mt * 16][kc * 32], 64, lane);
#pragma unroll
        for (int nt = 0; nt < 2; ++nt) oacc[mt][nt] = wmma_bf16(a, bf[nt], oacc[mt][nt]);
      }
    }
    __syncthreads();
  }

#pragma unroll
  for (int mt = 0; mt < 2; ++mt)
#pragma unroll
    for (int nt = 0; nt < 2; ++nt)
#pragma unroll
      for (int r = 0; r < 8; ++r) {
        int m = q0 + wr * 32 + mt * 16 + half * 8 + r;
        int d = (w & 3) * 32 + nt * 16 + ln;
        Ob[(size_t)m * QD + h * HD + d] = f2bf(oacc[mt][nt][r]);
      }
}

// ---------------------------------------------------------------------------
extern "C" void kernel_launch(void* const* d_in, const int* in_sizes, int n_in,
                              void* d_out, int out_size, void* d_ws, size_t ws_size,
                              hipStream_t stream) {
  const float* X  = (const float*)d_in[0];
  const int*   pos = (const int*)d_in[1];
  const float* Wq = (const float*)d_in[2];
  const float* Wk = (const float*)d_in[3];
  const float* Wv = (const float*)d_in[4];
  const float* Wo = (const float*)d_in[5];
  float* out = (float*)d_out;

  char* p = (char*)d_ws;
  auto alloc = [&](size_t bytes) {
    char* r = p;
    p += (bytes + 255) & ~(size_t)255;
    return r;
  };
  unsigned short* Xb  = (unsigned short*)alloc((size_t)S_LEN * HID_D * 2);
  unsigned short* Wqb = (unsigned short*)alloc((size_t)HID_D * QD * 2);
  unsigned short* Wkb = (unsigned short*)alloc((size_t)HID_D * KVD * 2);
  unsigned short* Wvb = (unsigned short*)alloc((size_t)HID_D * KVD * 2);
  unsigned short* Wob = (unsigned short*)alloc((size_t)QD * HID_D * 2);
  float* Qf = (float*)alloc((size_t)S_LEN * QD * 4);
  float* Kf = (float*)alloc((size_t)S_LEN * KVD * 4);
  float* Vf = (float*)alloc((size_t)S_LEN * KVD * 4);
  unsigned short* Qbb = (unsigned short*)alloc((size_t)S_LEN * QD * 2);
  unsigned short* Kbb = (unsigned short*)alloc((size_t)S_LEN * KVD * 2);
  unsigned short* Vbb = (unsigned short*)alloc((size_t)S_LEN * KVD * 2);
  unsigned short* Ob  = (unsigned short*)alloc((size_t)S_LEN * QD * 2);
  float* colpart = (float*)alloc((size_t)(S_LEN / 64) * NH * S_LEN * 4);
  float* colsum  = (float*)alloc((size_t)NH * S_LEN * 4);
  float* thr     = (float*)alloc(NH * 4);

  // 1) bf16 converts
  cvt_bf16_kernel<<<(S_LEN * HID_D / 4 + 255) / 256, 256, 0, stream>>>(X, Xb, S_LEN * HID_D / 4);
  cvt_bf16_kernel<<<(HID_D * QD / 4 + 255) / 256, 256, 0, stream>>>(Wq, Wqb, HID_D * QD / 4);
  cvt_bf16_kernel<<<(HID_D * KVD / 4 + 255) / 256, 256, 0, stream>>>(Wk, Wkb, HID_D * KVD / 4);
  cvt_bf16_kernel<<<(HID_D * KVD / 4 + 255) / 256, 256, 0, stream>>>(Wv, Wvb, HID_D * KVD / 4);
  cvt_bf16_kernel<<<(QD * HID_D / 4 + 255) / 256, 256, 0, stream>>>(Wo, Wob, QD * HID_D / 4);

  // 2) QKV projections (WMMA)
  gemm_bf16_kernel<<<dim3(QD / 128, S_LEN / 128), 256, 0, stream>>>(Xb, Wqb, Qf, S_LEN, QD, HID_D);
  gemm_bf16_kernel<<<dim3(KVD / 128, S_LEN / 128), 256, 0, stream>>>(Xb, Wkb, Kf, S_LEN, KVD, HID_D);
  gemm_bf16_kernel<<<dim3(KVD / 128, S_LEN / 128), 256, 0, stream>>>(Xb, Wvb, Vf, S_LEN, KVD, HID_D);

  // 3) RoPE (+bf16) for Q, K; plain convert for V
  rope_cvt_kernel<<<(S_LEN * NH * 64) / 256, 256, 0, stream>>>(Qf, pos, Qbb, NH);
  rope_cvt_kernel<<<(S_LEN * NKV * 64) / 256, 256, 0, stream>>>(Kf, pos, Kbb, NKV);
  cvt_bf16_kernel<<<(S_LEN * KVD / 4 + 255) / 256, 256, 0, stream>>>(Vf, Vbb, S_LEN * KVD / 4);

  // 4) first-softmax stats + column sums
  attn_stats_kernel<<<dim3(S_LEN / 64, NH), 256, 0, stream>>>(Qbb, Kbb, colpart);
  colsum_reduce_kernel<<<NH * S_LEN / 256, 256, 0, stream>>>(colpart, colsum);

  // 5) heavy-hitter threshold per head
  topk_thresh_kernel<<<NH, 256, 0, stream>>>(colsum, thr);

  // 6) masked attention + P@V
  attn_out_kernel<<<dim3(S_LEN / 64, NH), 256, 0, stream>>>(Qbb, Kbb, Vbb, colsum, thr, Ob);

  // 7) output projection (WMMA) -> f32 out
  gemm_bf16_kernel<<<dim3(HID_D / 128, S_LEN / 128), 256, 0, stream>>>(Ob, Wob, out, S_LEN, HID_D, QD);
}
